// DecoderNCP_56289841382059
// MI455X (gfx1250) — compile-verified, matrix operand order checked by reading, not run on hardware
//
#include <hip/hip_runtime.h>
#include <hip/hip_bf16.h>

typedef __bf16 bf16_t;
typedef __attribute__((ext_vector_type(16))) __bf16 v16bf;
typedef __attribute__((ext_vector_type(8)))  __bf16 v8bf;
typedef __attribute__((ext_vector_type(8)))  float  v8f;

// ---------------- problem constants ----------------
#define BATCH 512
#define TLEN  512
// layer dims
#define INTER  231
#define CMD    153
#define LATENT 128
// padded K per layer (multiple of 32)
#define PK0 256   // cat0 = 1 + 231 = 232
#define PK1 384   // cat1 = 231 + 153 = 384
#define PK2 288   // cat2 = 153 + 128 = 281
#define KT0 8
#define KT1 12
#define KT2 9
#define NT0 15    // ceil(231/16)
#define NT1 10    // ceil(153/16)
#define NT2 8     // 128/16
#define NWAVES 8
#define STAGE_STRIDE 240

// packed weight fragment sizes (bf16 elements): NT*KT*512
#define SZ0 (NT0*KT0*512)   // 61440
#define SZ1 (NT1*KT1*512)   // 61440
#define SZ2 (NT2*KT2*512)   // 36864

// ---------------- weight pre-pack (f32 -> bf16, WMMA B-frag layout) ----------------
// frag element (nt,kt,lane,j): lane L covers W row n = nt*16 + (L&15);
// h = L>>4; j<8 -> k = kt*32 + h*8 + j ; j>=8 -> k = kt*32 + 16 + h*8 + (j-8)
__global__ void cfc_pack_w(const float* __restrict__ W, const float* __restrict__ mask,
                           bf16_t* __restrict__ dst, int OUT, int CAT, int NT, int KT) {
    int total = NT * KT * 512;
    for (int idx = blockIdx.x * blockDim.x + threadIdx.x; idx < total;
         idx += gridDim.x * blockDim.x) {
        int j    = idx & 15;
        int lane = (idx >> 4) & 31;
        int tk   = idx >> 9;
        int kt   = tk % KT;
        int nt   = tk / KT;
        int n    = nt * 16 + (lane & 15);
        int h    = lane >> 4;
        int k    = kt * 32 + ((j < 8) ? (h * 8 + j) : (16 + h * 8 + (j - 8)));
        float v = 0.0f;
        if (n < OUT && k < CAT) {
            v = W[n * CAT + k];
            if (mask) v *= mask[n * CAT + k];
        }
        dst[idx] = (bf16_t)v;
    }
}

// ---------------- fragment loaders ----------------
__device__ __forceinline__ v16bf load_a_lds(const bf16_t* xb, int kstride, int kt, int lane) {
    int m = lane & 15, h = lane >> 4;
    const bf16_t* p = xb + m * kstride + kt * 32 + h * 8;
    v8bf lo = *(const v8bf*)(p);
    v8bf hi = *(const v8bf*)(p + 16);
    v16bf r;
#pragma unroll
    for (int i = 0; i < 8; ++i) { r[i] = lo[i]; r[8 + i] = hi[i]; }
    return r;
}

__device__ __forceinline__ v16bf load_b_frag(const bf16_t* p) {
    v8bf lo = *(const v8bf*)(p);
    v8bf hi = *(const v8bf*)(p + 8);
    v16bf r;
#pragma unroll
    for (int i = 0; i < 8; ++i) { r[i] = lo[i]; r[8 + i] = hi[i]; }
    return r;
}

__device__ __forceinline__ v8f wmma_bf16(v16bf a, v16bf b, v8f c) {
    return __builtin_amdgcn_wmma_f32_16x16x32_bf16(false, a, false, b, (short)0, c,
                                                   false, false);
}

// ---------------- one CfC cell for a 16-row batch tile ----------------
// out[m][n] = interp(tanh(x@Wf1.T+b1), tanh(x@Wf2.T+b2), sigmoid(x@(Wa+Wb).T + ba+bb))
__device__ __forceinline__ void layer_compute(
    const bf16_t* __restrict__ xb, int kstride, int KT,
    const bf16_t* __restrict__ Wf1, const bf16_t* __restrict__ Wf2,
    const bf16_t* __restrict__ Wta, const bf16_t* __restrict__ Wtb,
    const float* __restrict__ b1, const float* __restrict__ b2,
    const float* __restrict__ ba, const float* __restrict__ bb,
    int NT, int OUT, float* __restrict__ stage, int lane, int wave) {
    for (int nt = wave; nt < NT; nt += NWAVES) {
        v8f acc1 = {}, acc2 = {}, acct = {};
        for (int kt = 0; kt < KT; ++kt) {
            v16bf A = load_a_lds(xb, kstride, kt, lane);
            const int fo = ((nt * KT + kt) << 9) + lane * 16;
            __builtin_prefetch(Wf1 + fo + 512, 0, 0);   // next K-tile weight stream
            acc1 = wmma_bf16(A, load_b_frag(Wf1 + fo), acc1);
            acc2 = wmma_bf16(A, load_b_frag(Wf2 + fo), acc2);
            acct = wmma_bf16(A, load_b_frag(Wta + fo), acct);
            acct = wmma_bf16(A, load_b_frag(Wtb + fo), acct);
        }
        int n  = nt * 16 + (lane & 15);
        int ns = (n < OUT) ? n : (OUT - 1);
        float bia1 = b1[ns], bia2 = b2[ns], biat = ba[ns] + bb[ns];
        int mbase = (lane >> 4) * 8;   // C/D layout: VGPR r holds M=r (lanes 0-15), M=8+r (lanes 16-31)
#pragma unroll
        for (int r = 0; r < 8; ++r) {
            float f1 = tanhf(acc1[r] + bia1);
            float f2 = tanhf(acc2[r] + bia2);
            float tv = 1.0f / (1.0f + __expf(-(acct[r] + biat)));
            stage[(mbase + r) * STAGE_STRIDE + n] = f1 * (1.0f - tv) + tv * f2;
        }
    }
}

// ---------------- main persistent-recurrence kernel ----------------
__global__ void __launch_bounds__(256)
cfc_rnn_kernel(const float* __restrict__ elapsed, const float* __restrict__ enc,
               const bf16_t* __restrict__ wp,
               const float* __restrict__ b1_0, const float* __restrict__ b2_0,
               const float* __restrict__ ba_0, const float* __restrict__ bb_0,
               const float* __restrict__ b1_1, const float* __restrict__ b2_1,
               const float* __restrict__ ba_1, const float* __restrict__ bb_1,
               const float* __restrict__ b1_2, const float* __restrict__ b2_2,
               const float* __restrict__ ba_2, const float* __restrict__ bb_2,
               float* __restrict__ out_rnn, float* __restrict__ out_hidden) {
    __shared__ __align__(16) bf16_t xb0[16 * PK0];
    __shared__ __align__(16) bf16_t xb1[16 * PK1];
    __shared__ __align__(16) bf16_t xb2[16 * PK2];
    __shared__ float hS[16 * 512];
    __shared__ float stage[16 * STAGE_STRIDE];

    const int tid  = threadIdx.x;
    const int lane = tid & 31;
    const int wave = tid >> 5;
    const int b0   = blockIdx.x * 16;

    // packed weight bases
    const bf16_t* Wf1_0 = wp;
    const bf16_t* Wf2_0 = wp + SZ0;
    const bf16_t* Wta_0 = wp + 2 * SZ0;
    const bf16_t* Wtb_0 = wp + 3 * SZ0;
    const bf16_t* Wf1_1 = wp + 4 * SZ0;
    const bf16_t* Wf2_1 = Wf1_1 + SZ1;
    const bf16_t* Wta_1 = Wf1_1 + 2 * SZ1;
    const bf16_t* Wtb_1 = Wf1_1 + 3 * SZ1;
    const bf16_t* Wf1_2 = Wf1_1 + 4 * SZ1;
    const bf16_t* Wf2_2 = Wf1_2 + SZ2;
    const bf16_t* Wta_2 = Wf1_2 + 2 * SZ2;
    const bf16_t* Wtb_2 = Wf1_2 + 3 * SZ2;

    // ---- init: zero pads, load encoder hidden state ----
    for (int i = tid; i < 16 * PK0; i += 256) xb0[i] = (bf16_t)0.0f;
    for (int i = tid; i < 16 * PK2; i += 256) xb2[i] = (bf16_t)0.0f;
    __syncthreads();
    for (int i = tid; i < 16 * 512; i += 256) {
        int m = i >> 9, c = i & 511;
        float v = enc[(size_t)(b0 + m) * 512 + c];
        hS[i] = v;
        bf16_t bv = (bf16_t)v;
        if (c < INTER)            xb0[m * PK0 + 1 + c] = bv;            // h0
        else if (c < INTER + CMD) xb1[m * PK1 + INTER + (c - INTER)] = bv;   // h1
        else                      xb2[m * PK2 + CMD + (c - INTER - CMD)] = bv; // h2
    }
    __syncthreads();

    // ---- recurrence over time ----
    for (int t = 0; t < TLEN; ++t) {
        if (tid < 16) xb0[tid * PK0] = (bf16_t)elapsed[(size_t)(b0 + tid) * TLEN + t];
        __syncthreads();

        // layer 0: (x_t, h0) -> n0
        layer_compute(xb0, PK0, KT0, Wf1_0, Wf2_0, Wta_0, Wtb_0,
                      b1_0, b2_0, ba_0, bb_0, NT0, INTER, stage, lane, wave);
        __syncthreads();
        for (int i = tid; i < 16 * INTER; i += 256) {
            int m = i / INTER, n = i - m * INTER;
            float v = stage[m * STAGE_STRIDE + n];
            bf16_t bv = (bf16_t)v;
            xb1[m * PK1 + n]     = bv;   // input to layer 1
            xb0[m * PK0 + 1 + n] = bv;   // h0 for next step
            hS[m * 512 + n]      = v;
        }
        __syncthreads();

        // layer 1: (n0, h1) -> n1
        layer_compute(xb1, PK1, KT1, Wf1_1, Wf2_1, Wta_1, Wtb_1,
                      b1_1, b2_1, ba_1, bb_1, NT1, CMD, stage, lane, wave);
        __syncthreads();
        for (int i = tid; i < 16 * CMD; i += 256) {
            int m = i / CMD, n = i - m * CMD;
            float v = stage[m * STAGE_STRIDE + n];
            bf16_t bv = (bf16_t)v;
            xb2[m * PK2 + n]         = bv;   // input to layer 2
            xb1[m * PK1 + INTER + n] = bv;   // h1 for next step
            hS[m * 512 + INTER + n]  = v;
        }
        __syncthreads();

        // layer 2: (n1, h2) -> n2
        layer_compute(xb2, PK2, KT2, Wf1_2, Wf2_2, Wta_2, Wtb_2,
                      b1_2, b2_2, ba_2, bb_2, NT2, LATENT, stage, lane, wave);
        __syncthreads();
        for (int i = tid; i < 16 * LATENT; i += 256) {
            int m = i >> 7, n = i & 127;
            float v = stage[m * STAGE_STRIDE + n];
            xb2[m * PK2 + CMD + n]        = (bf16_t)v;   // h2 for next step
            hS[m * 512 + INTER + CMD + n] = v;
            out_rnn[((size_t)(b0 + m) * TLEN + t) * LATENT + n] = v;
        }
        __syncthreads();
    }

    // ---- final hidden state ----
    for (int i = tid; i < 16 * 512; i += 256) {
        out_hidden[(size_t)(b0 + (i >> 9)) * 512 + (i & 511)] = hS[i];
    }
}

// ---------------- host launcher ----------------
extern "C" void kernel_launch(void* const* d_in, const int* in_sizes, int n_in,
                              void* d_out, int out_size, void* d_ws, size_t ws_size,
                              hipStream_t stream) {
    const float* elapsed = (const float*)d_in[0];
    const float* enc     = (const float*)d_in[1];

    const float* W[3][4];
    const float* bias[3][4];
    const float* mask[3];
    int base = 2;
    for (int l = 0; l < 3; ++l) {
        for (int j = 0; j < 4; ++j) {
            W[l][j]    = (const float*)d_in[base + 2 * j];
            bias[l][j] = (const float*)d_in[base + 2 * j + 1];
        }
        mask[l] = (const float*)d_in[base + 8];
        base += 9;
    }

    bf16_t* wp = (bf16_t*)d_ws;
    const int OUTd[3] = {INTER, CMD, LATENT};
    const int CATd[3] = {232, 384, 281};
    const int NTd[3]  = {NT0, NT1, NT2};
    const int KTd[3]  = {KT0, KT1, KT2};
    const int SZd[3]  = {SZ0, SZ1, SZ2};

    size_t off = 0;
    for (int l = 0; l < 3; ++l) {
        for (int j = 0; j < 4; ++j) {
            int total = SZd[l];
            int blocks = (total + 255) / 256;
            const float* mk = (j < 2) ? mask[l] : nullptr;   // ff1/ff2 masked; ta/tb dense
            cfc_pack_w<<<blocks, 256, 0, stream>>>(W[l][j], mk, wp + off,
                                                   OUTd[l], CATd[l], NTd[l], KTd[l]);
            off += (size_t)total;
        }
    }

    float* out_rnn    = (float*)d_out;
    float* out_hidden = out_rnn + (size_t)BATCH * TLEN * LATENT;

    cfc_rnn_kernel<<<BATCH / 16, 256, 0, stream>>>(
        elapsed, enc, wp,
        bias[0][0], bias[0][1], bias[0][2], bias[0][3],
        bias[1][0], bias[1][1], bias[1][2], bias[1][3],
        bias[2][0], bias[2][1], bias[2][2], bias[2][3],
        out_rnn, out_hidden);
}
